// GNMTDecoder_22110491640159
// MI455X (gfx1250) — matrix operand order repaired and checked by reading, not compiled, not run on hardware
//
#include <hip/hip_runtime.h>

// ---------------------------------------------------------------------------
// GNMT decoder for MI455X (gfx1250): bf16 WMMA GEMMs, fp32 accumulate.
// Weights converted fp32->bf16 once per launch (w_out bf16 = 131MB fits 192MB L2,
// so the dominant per-step vocab GEMM streams from L2 instead of HBM).
// ---------------------------------------------------------------------------

typedef __bf16 bf16;
typedef __attribute__((ext_vector_type(16))) __bf16 v16bf;
typedef __attribute__((ext_vector_type(8)))  __bf16 v8bf;
typedef __attribute__((ext_vector_type(8)))  float  v8f;
typedef __attribute__((ext_vector_type(4)))  unsigned int v4u;
typedef __attribute__((ext_vector_type(8)))  int v8i;
typedef __attribute__((ext_vector_type(4)))  int v4i;

#if defined(__has_builtin)
#  if __has_builtin(__builtin_amdgcn_tensor_load_to_lds)
#    define HAVE_TDM 1
#  endif
#  if __has_builtin(__builtin_amdgcn_s_wait_tensorcnt)
#    define HAVE_TCNT 1
#  endif
#endif

constexpr int VOC = 32000;
constexpr int EMB = 1024;
constexpr int HID = 1024;
constexpr int NL  = 4;
constexpr int NB  = 32;
constexpr int NT  = 64;
constexpr int NS  = 64;
constexpr int GH4 = 4 * HID;   // 4096 gate width

// ----------------------------- helpers -------------------------------------

__device__ __forceinline__ bf16 f2bf(float f) {
  unsigned u = __builtin_bit_cast(unsigned, f);
  unsigned r = (u + 0x7FFFu + ((u >> 16) & 1u)) >> 16;     // round-nearest-even
  unsigned short s = (unsigned short)r;
  return __builtin_bit_cast(bf16, s);
}

__device__ __forceinline__ float sigf(float x) { return 1.0f / (1.0f + expf(-x)); }

// A fragment: 16x32 bf16, row-major A (lda = K). ISA 7.12.2 16-bit A layout:
// lanes 0-15 hold row M=lane, K = {kb+0..7, kb+16..23}; lanes 16-31 hold
// K = {kb+8..15, kb+24..31}.  Two b128 loads per lane.
__device__ __forceinline__ v16bf load_a(const bf16* __restrict__ A, int lda,
                                        int m0, int kb, int lane) {
  int row = lane & 15, hi = lane >> 4;
  const bf16* p = A + (size_t)(m0 + row) * lda + kb + hi * 8;
  v8bf l0 = *(const v8bf*)p;
  v8bf l1 = *(const v8bf*)(p + 16);
  v16bf r;
#pragma unroll
  for (int i = 0; i < 8; ++i) { r[i] = l0[i]; r[i + 8] = l1[i]; }
  return r;
}

// B fragment: 32x16 bf16 where B = W^T, W stored row-major (N x K, ldb = K).
// lanes 0-15 hold column N=lane, K = {kb+0..7, kb+16..23}; lanes 16-31 hold K = {kb+8..15, kb+24..31}.
__device__ __forceinline__ v16bf load_b(const bf16* __restrict__ W, int ldb,
                                        int n0, int kb, int lane) {
  int col = lane & 15, hi = lane >> 4;
  const bf16* p = W + (size_t)(n0 + col) * ldb + kb + hi * 8;
  v8bf l0 = *(const v8bf*)p;
  v8bf l1 = *(const v8bf*)(p + 16);
  v16bf r;
#pragma unroll
  for (int i = 0; i < 8; ++i) { r[i] = l0[i]; r[i + 8] = l1[i]; }
  return r;
}

// ----------------------------- kernels -------------------------------------

__global__ void cvt_f32_bf16(const float* __restrict__ src, bf16* __restrict__ dst, int n) {
  int i = blockIdx.x * blockDim.x + threadIdx.x;
  if (i < n) dst[i] = f2bf(src[i]);
}

// emb_bf layout (T, B, EM), padding_idx = 0 zeroed.
__global__ void embed_gather(const int* __restrict__ toks, const float* __restrict__ embed,
                             bf16* __restrict__ emb_bf) {
  int i = blockIdx.x * blockDim.x + threadIdx.x;
  if (i >= NT * NB * EMB) return;
  int e = i % EMB;
  int bt = i / EMB;
  int b = bt % NB;
  int t = bt / NB;
  int tok = toks[b * NT + t];
  float v = (tok == 0) ? 0.0f : embed[(size_t)tok * EMB + e];
  emb_bf[i] = f2bf(v);
}

__global__ void init_state(float* __restrict__ h, float* __restrict__ c,
                           bf16* __restrict__ hbf, float* __restrict__ ctx) {
  int i = blockIdx.x * blockDim.x + threadIdx.x;
  if (i < NL * NB * HID) { h[i] = 0.0f; c[i] = 0.0f; hbf[i] = f2bf(0.0f); }
  if (i < NB * HID) ctx[i] = 0.0f;
}

// x0 = concat(emb_t, ctx) in bf16, shape (B, EM+H)
__global__ void build_x0(const bf16* __restrict__ emb_t, const float* __restrict__ ctx,
                         bf16* __restrict__ x0) {
  int i = blockIdx.x * blockDim.x + threadIdx.x;
  if (i >= NB * (EMB + HID)) return;
  int b = i / (EMB + HID), j = i % (EMB + HID);
  x0[i] = (j < EMB) ? emb_t[(size_t)b * EMB + j]
                    : f2bf(ctx[(size_t)b * HID + (j - EMB)]);
}

// out[M,N] (fp32) = A1 @ W1^T (+ A2 @ W2^T) + bias1 + bias2.
// A row-major (M,K), W row-major (N,K). One wave per 16x16 tile.
__global__ __launch_bounds__(256)
void wmma_gemm(const bf16* __restrict__ A1, const bf16* __restrict__ W1, int K1,
               const bf16* __restrict__ A2, const bf16* __restrict__ W2, int K2,
               const float* __restrict__ bias1, const float* __restrict__ bias2,
               float* __restrict__ outp, long long ldc, int M, int N, int do_tdm) {
  const int lane = threadIdx.x & 31;
  const int wave = blockIdx.x * (blockDim.x >> 5) + (threadIdx.x >> 5);
  const int tn = N >> 4;
  const int tiles = (M >> 4) * tn;

#if defined(HAVE_TDM)
  // TDM prefetch of the weight tensor's leading tile into LDS (async, wave 0):
  // 2D descriptor: tensor (N x K1) bf16, tile (2 x K1).
  if (do_tdm) {
    if (wave == 0) {
      unsigned long long ga = (unsigned long long)W1;
      v4u g0; v8i g1; v4i g2; v4i g3; v8i g4;
      g0[0] = 1u;                                            // count=1, user mode
      g0[1] = 0u;                                            // lds_addr = base
      g0[2] = (unsigned)(ga & 0xFFFFFFFFull);                // global_addr lo
      g0[3] = (unsigned)((ga >> 32) & 0x1FFFFFFull) | (2u << 30);  // hi | type=2
      g1[0] = (int)(1u << 16);                               // data_size=2B
      g1[1] = (int)(((unsigned)K1 & 0xFFFFu) << 16);         // tensor_dim0 lo
      g1[2] = (int)((((unsigned)K1 >> 16) & 0xFFFFu) |
                    (((unsigned)N & 0xFFFFu) << 16));        // dim0 hi | dim1 lo
      g1[3] = (int)((((unsigned)N >> 16) & 0xFFFFu) |
                    (((unsigned)K1 & 0xFFFFu) << 16));       // dim1 hi | tile_dim0
      g1[4] = 2;                                             // tile_dim1 = 2 rows
      g1[5] = (int)(unsigned)K1;                             // dim0_stride lo32
      g1[6] = (int)(((unsigned)K1 & 0xFFFFu) << 16);         // stride hi | dim1_stride lo
      g1[7] = 0;
      g2[0] = g2[1] = g2[2] = g2[3] = 0;
      g3[0] = g3[1] = g3[2] = g3[3] = 0;
      g4[0] = g4[1] = g4[2] = g4[3] = 0;
      g4[4] = g4[5] = g4[6] = g4[7] = 0;
      // 6-arg form (upstream clang-23 / therock-10.0 headers)
      __builtin_amdgcn_tensor_load_to_lds(g0, g1, g2, g3, g4, 0);
    }
#if defined(HAVE_TCNT)
    __builtin_amdgcn_s_wait_tensorcnt(0);
#endif
  }
#endif

  if (wave >= tiles) return;          // wave-uniform: EXEC all-1s for WMMA
  const int m0 = (wave / tn) << 4;
  const int n0 = (wave % tn) << 4;

  __builtin_prefetch(W1 + (size_t)n0 * K1, 0, 0);   // global_prefetch_b8

  v8f acc = {0.f, 0.f, 0.f, 0.f, 0.f, 0.f, 0.f, 0.f};

  for (int k = 0; k < K1; k += 32) {
    v16bf a = load_a(A1, K1, m0, k, lane);
    v16bf b = load_b(W1, K1, n0, k, lane);
    acc = __builtin_amdgcn_wmma_f32_16x16x32_bf16(false, a, false, b,
                                                  (short)0, acc, false, false);
  }
  if (A2 != nullptr) {
    for (int k = 0; k < K2; k += 32) {
      v16bf a = load_a(A2, K2, m0, k, lane);
      v16bf b = load_b(W2, K2, n0, k, lane);
      acc = __builtin_amdgcn_wmma_f32_16x16x32_bf16(false, a, false, b,
                                                    (short)0, acc, false, false);
    }
  }

  // C/D layout: VGPR r -> M = r + 8*(lane>=16), N = lane & 15
  const int nc = lane & 15;
  const int ms = (lane >> 4) * 8;
  float badd = 0.0f;
  if (bias1) badd += bias1[n0 + nc];
  if (bias2) badd += bias2[n0 + nc];
#pragma unroll
  for (int r = 0; r < 8; ++r) {
    int m = m0 + ms + r;
    outp[(size_t)m * (size_t)ldc + n0 + nc] = acc[r] + badd;
  }
}

// Elementwise LSTM cell over (B, H); gate order i,f,g,o along 4H.
__global__ void lstm_cell(const float* __restrict__ gates, const float* __restrict__ x_in,
                          float* __restrict__ h, float* __restrict__ c,
                          float* __restrict__ x_out_f, bf16* __restrict__ x_out_bf,
                          bf16* __restrict__ h_bf, int residual) {
  int i = blockIdx.x * blockDim.x + threadIdx.x;
  if (i >= NB * HID) return;
  int b = i / HID, n = i % HID;
  const float* g = gates + (size_t)b * GH4;
  float gi = g[n], gf = g[HID + n], gg = g[2 * HID + n], go = g[3 * HID + n];
  float c2 = sigf(gf) * c[i] + sigf(gi) * tanhf(gg);
  float ht = sigf(go) * tanhf(c2);
  if (residual) ht += x_in[i];
  c[i] = c2;
  h[i] = ht;
  x_out_f[i] = ht;
  x_out_bf[i] = f2bf(ht);
  h_bf[i] = f2bf(ht);
}

// Per-batch attention: scores -> masked softmax -> context; also assembles
// bf16 concat(hl, ctx) for the vocab GEMM and writes attn weights to d_out.
__global__ void attention_step(const float* __restrict__ enc, const float* __restrict__ pq,
                               const unsigned char* __restrict__ mask,
                               const float* __restrict__ hl,
                               float* __restrict__ ctx, float* __restrict__ attn_out,
                               long long attn_ld, bf16* __restrict__ hlctx) {
  int b = blockIdx.x;
  int tid = threadIdx.x;
  __shared__ float sc[NS];

  for (int s = tid; s < NS; s += blockDim.x) {
    const float* er = enc + ((size_t)b * NS + s) * HID;
    const float* q  = pq + (size_t)b * HID;
    float a = 0.0f;
    for (int h = 0; h < HID; ++h) a += er[h] * q[h];
    sc[s] = mask[b * NS + s] ? -1e9f : a;
  }
  __syncthreads();
  if (tid == 0) {
    float mx = sc[0];
    for (int s = 1; s < NS; ++s) mx = fmaxf(mx, sc[s]);
    float sum = 0.0f;
    for (int s = 0; s < NS; ++s) { sc[s] = expf(sc[s] - mx); sum += sc[s]; }
    float inv = 1.0f / sum;
    for (int s = 0; s < NS; ++s) sc[s] *= inv;
  }
  __syncthreads();
  for (int s = tid; s < NS; s += blockDim.x)
    attn_out[(size_t)b * (size_t)attn_ld + s] = sc[s];
  for (int h = tid; h < HID; h += blockDim.x) {
    float a = 0.0f;
    for (int s = 0; s < NS; ++s) a += sc[s] * enc[((size_t)b * NS + s) * HID + h];
    ctx[(size_t)b * HID + h] = a;
    hlctx[(size_t)b * 2 * HID + h]        = f2bf(hl[(size_t)b * HID + h]);
    hlctx[(size_t)b * 2 * HID + HID + h]  = f2bf(a);
  }
}

// ----------------------------- host side -----------------------------------

extern "C" void kernel_launch(void* const* d_in, const int* in_sizes, int n_in,
                              void* d_out, int out_size, void* d_ws, size_t ws_size,
                              hipStream_t stream) {
  (void)in_sizes; (void)n_in; (void)out_size; (void)ws_size;

  const int*   tgt    = (const int*)d_in[0];
  const float* enc    = (const float*)d_in[1];
  const unsigned char* mask = (const unsigned char*)d_in[2];
  const float* embed  = (const float*)d_in[3];
  const float* w_ih0  = (const float*)d_in[4];
  const float* b_ih0  = (const float*)d_in[5];
  const float* w_ihr  = (const float*)d_in[6];
  const float* b_ihr  = (const float*)d_in[7];
  const float* w_hh   = (const float*)d_in[8];
  const float* b_hh   = (const float*)d_in[9];
  const float* w_attn = (const float*)d_in[10];
  const float* w_out  = (const float*)d_in[11];
  const float* b_out  = (const float*)d_in[12];

  float* out_logits = (float*)d_out;                         // (B, T, V)
  float* out_attn   = out_logits + (size_t)NB * NT * VOC;    // (B, T, S)

  char* ws = (char*)d_ws;
  size_t off = 0;
  auto alloc = [&](size_t bytes) {
    size_t r = off;
    off = (off + bytes + 255) & ~(size_t)255;
    return r;
  };
  bf16*  wih0_bf  = (bf16*)(ws + alloc((size_t)GH4 * (EMB + HID) * 2));
  bf16*  wihr_bf  = (bf16*)(ws + alloc((size_t)(NL - 1) * GH4 * HID * 2));
  bf16*  whh_bf   = (bf16*)(ws + alloc((size_t)NL * GH4 * HID * 2));
  bf16*  wattn_bf = (bf16*)(ws + alloc((size_t)HID * HID * 2));
  bf16*  wout_bf  = (bf16*)(ws + alloc((size_t)VOC * 2 * HID * 2));
  bf16*  emb_bf   = (bf16*)(ws + alloc((size_t)NT * NB * EMB * 2));
  float* hstate   = (float*)(ws + alloc((size_t)NL * NB * HID * 4));
  float* cstate   = (float*)(ws + alloc((size_t)NL * NB * HID * 4));
  bf16*  h_bf     = (bf16*) (ws + alloc((size_t)NL * NB * HID * 2));
  float* ctxf     = (float*)(ws + alloc((size_t)NB * HID * 4));
  float* gatesf   = (float*)(ws + alloc((size_t)NB * GH4 * 4));
  bf16*  x0_bf    = (bf16*) (ws + alloc((size_t)NB * (EMB + HID) * 2));
  bf16*  xl_bf    = (bf16*) (ws + alloc((size_t)NB * HID * 2));
  float* xlf      = (float*)(ws + alloc((size_t)NB * HID * 4));
  float* pqf      = (float*)(ws + alloc((size_t)NB * HID * 4));
  bf16*  hlctx_bf = (bf16*) (ws + alloc((size_t)NB * 2 * HID * 2));

  auto cvt = [&](const float* s, bf16* d, long long n) {
    cvt_f32_bf16<<<(int)((n + 255) / 256), 256, 0, stream>>>(s, d, (int)n);
  };
  // one-time (per launch) weight down-conversion
  cvt(w_ih0,  wih0_bf,  (long long)GH4 * (EMB + HID));
  cvt(w_ihr,  wihr_bf,  (long long)(NL - 1) * GH4 * HID);
  cvt(w_hh,   whh_bf,   (long long)NL * GH4 * HID);
  cvt(w_attn, wattn_bf, (long long)HID * HID);
  cvt(w_out,  wout_bf,  (long long)VOC * 2 * HID);
  embed_gather<<<(NT * NB * EMB + 255) / 256, 256, 0, stream>>>(tgt, embed, emb_bf);
  init_state<<<(NL * NB * HID + 255) / 256, 256, 0, stream>>>(hstate, cstate, h_bf, ctxf);

  auto gemm = [&](const bf16* A1, const bf16* W1, int K1,
                  const bf16* A2, const bf16* W2, int K2,
                  const float* B1, const float* B2,
                  float* outp, long long ldc, int M, int N, int tdm) {
    int tiles = (M / 16) * (N / 16);
    int blocks = (tiles + 7) / 8;   // 8 waves (16x16 tiles) per 256-thread block
    wmma_gemm<<<blocks, 256, 0, stream>>>(A1, W1, K1, A2, W2, K2, B1, B2,
                                          outp, ldc, M, N, tdm);
  };

  for (int t = 0; t < NT; ++t) {
    const bf16* emb_t = emb_bf + (size_t)t * NB * EMB;
    build_x0<<<(NB * (EMB + HID) + 255) / 256, 256, 0, stream>>>(emb_t, ctxf, x0_bf);

    for (int l = 0; l < NL; ++l) {
      const bf16*  A1 = (l == 0) ? x0_bf : xl_bf;
      const int    K1 = (l == 0) ? (EMB + HID) : HID;
      const bf16*  W1 = (l == 0) ? wih0_bf : wihr_bf + (size_t)(l - 1) * GH4 * HID;
      const float* B1 = (l == 0) ? b_ih0 : b_ihr + (size_t)(l - 1) * GH4;
      const bf16*  A2 = h_bf + (size_t)l * NB * HID;
      const bf16*  W2 = whh_bf + (size_t)l * GH4 * HID;
      const float* B2 = b_hh + (size_t)l * GH4;
      gemm(A1, W1, K1, A2, W2, HID, B1, B2, gatesf, GH4, NB, GH4, 0);
      lstm_cell<<<(NB * HID + 255) / 256, 256, 0, stream>>>(
          gatesf, xlf,
          hstate + (size_t)l * NB * HID, cstate + (size_t)l * NB * HID,
          xlf, xl_bf, h_bf + (size_t)l * NB * HID, (l >= 2) ? 1 : 0);
    }

    // attention query: pq = hl @ w_attn^T
    gemm(xl_bf, wattn_bf, HID, nullptr, nullptr, 0, nullptr, nullptr,
         pqf, HID, NB, HID, 0);
    attention_step<<<NB, 256, 0, stream>>>(
        enc, pqf, mask, hstate + (size_t)3 * NB * HID,
        ctxf, out_attn + (size_t)t * NS, (long long)NT * NS, hlctx_bf);

    // vocab projection straight into d_out (bias fused); TDM prefetch enabled
    gemm(hlctx_bf, wout_bf, 2 * HID, nullptr, nullptr, 0, b_out, nullptr,
         out_logits + (size_t)t * VOC, (long long)NT * VOC, NB, VOC, 1);
  }
}
